// TransLayer_57801669870040
// MI455X (gfx1250) — compile-verified, matrix operand order checked by reading, not run on hardware
//
#include <hip/hip_runtime.h>

// Nystromformer layer for gfx1250 (MI455X), bf16 WMMA + fp32 accumulate.
#define B_   4
#define N_   4096
#define D_   512
#define H_   8
#define DH_  64
#define M_   256
#define BH_  32
#define NT_  16384
#define KS_  33

typedef __bf16 bf16;
typedef __attribute__((ext_vector_type(16))) __bf16 v16bf;
typedef __attribute__((ext_vector_type(8)))  __bf16 v8bf;
typedef __attribute__((ext_vector_type(8)))  float  v8f;

__device__ __forceinline__ v8f wmma_bf(v16bf a, v16bf b, v8f c) {
  return __builtin_amdgcn_wmma_f32_16x16x32_bf16(false, a, false, b, (short)0, c,
                                                 false, false);
}

// Load a 16x32 bf16 A/B fragment (NT form: rows K-contiguous).
// Lane L: row = row0 + L%16, half = L/16; k-chunks [half*8,+8) and [16+half*8,+8).
__device__ __forceinline__ v16bf ld_frag(const bf16* base, int ld, int row0, int k0) {
  int lane = threadIdx.x & 31;
  long r   = row0 + (lane & 15);
  int  h   = (lane >> 4) & 1;
  const bf16* p = base + r * (long)ld + k0 + h * 8;
  v8bf lo = *(const v8bf*)p;
  v8bf hi = *(const v8bf*)(p + 16);
  v16bf f;
#pragma unroll
  for (int i = 0; i < 8; ++i) { f[i] = lo[i]; f[i + 8] = hi[i]; }
  return f;
}

// ---------------- LayerNorm + cast to bf16 ----------------
__global__ void k_ln(const float* __restrict__ x, const float* __restrict__ gamma,
                     const float* __restrict__ beta, bf16* __restrict__ xn) {
  __shared__ float red[256];
  long row = blockIdx.x;
  int  t   = threadIdx.x;
  const float* xr = x + row * (long)D_;
  float a = xr[t], b = xr[t + 256];
  red[t] = a + b;
  __syncthreads();
  for (int off = 128; off > 0; off >>= 1) {
    if (t < off) red[t] += red[t + off];
    __syncthreads();
  }
  float mu = red[0] * (1.0f / D_);
  __syncthreads();
  float da = a - mu, db = b - mu;
  red[t] = da * da + db * db;
  __syncthreads();
  for (int off = 128; off > 0; off >>= 1) {
    if (t < off) red[t] += red[t + off];
    __syncthreads();
  }
  float inv = rsqrtf(red[0] * (1.0f / D_) + 1e-5f);
  xn[row * D_ + t]       = (bf16)(da * inv * gamma[t] + beta[t]);
  xn[row * D_ + t + 256] = (bf16)(db * inv * gamma[t + 256] + beta[t + 256]);
}

// ---------------- transpose + cast weights ----------------
__global__ void k_prepw(const float* __restrict__ wqkv, const float* __restrict__ wout,
                        bf16* __restrict__ wqkvT, bf16* __restrict__ woutT) {
  long i  = (long)blockIdx.x * blockDim.x + threadIdx.x;
  long n1 = (long)D_ * 1536;
  if (i < n1) {
    long r = i / 1536, c = i % 1536;
    wqkvT[c * D_ + r] = (bf16)wqkv[i];
  } else {
    long j = i - n1;
    if (j < (long)D_ * D_) {
      long r = j / D_, c = j % D_;
      woutT[c * D_ + r] = (bf16)wout[j];
    }
  }
}

// ---------------- qkv GEMM (scatter epilogue) ----------------
__global__ void k_qkv(const bf16* __restrict__ xn, const bf16* __restrict__ wT,
                      bf16* __restrict__ qb, bf16* __restrict__ kb,
                      bf16* __restrict__ vb, bf16* __restrict__ vT) {
  int row0 = blockIdx.y * 16;
  int col0 = blockIdx.x * 64;
  v8f acc[4] = {};
  for (int ks = 0; ks < 16; ++ks) {
    int k0 = ks * 32;
    v16bf af = ld_frag(xn, D_, row0, k0);
#pragma unroll
    for (int t = 0; t < 4; ++t)
      acc[t] = wmma_bf(af, ld_frag(wT, D_, col0 + t * 16, k0), acc[t]);
  }
  int lane = threadIdx.x & 31, c = lane & 15, hf = lane >> 4;
#pragma unroll
  for (int t = 0; t < 4; ++t) {
    int col = col0 + t * 16 + c;
    int which = col >> 9, hh = (col >> 6) & 7, dh = col & 63;
#pragma unroll
    for (int v = 0; v < 8; ++v) {
      int tok = row0 + hf * 8 + v;
      int bi = tok >> 12, ni = tok & 4095;
      int bh = bi * 8 + hh;
      long o = (((long)bh << 12) + ni) * 64 + dh;
      float val = acc[t][v];
      if (which == 0)      qb[o] = (bf16)(val * 0.125f);   // 1/sqrt(64)
      else if (which == 1) kb[o] = (bf16)val;
      else { vb[o] = (bf16)val; vT[(((long)bh << 6) + dh) * 4096 + ni] = (bf16)val; }
    }
  }
}

// ---------------- landmark means ----------------
__global__ void k_landmark(const bf16* __restrict__ qb, const bf16* __restrict__ kb,
                           bf16* __restrict__ ql, bf16* __restrict__ kl) {
  long idx = (long)blockIdx.x * blockDim.x + threadIdx.x;
  if (idx >= (long)BH_ * M_ * DH_) return;
  int dh = idx & 63;
  long r = idx >> 6;
  int mi = r & 255, bh = (int)(r >> 8);
  long base = (((long)bh << 12) + mi * 16) * 64 + dh;
  float sq = 0.f, sk = 0.f;
#pragma unroll
  for (int j = 0; j < 16; ++j) { sq += (float)qb[base + j * 64]; sk += (float)kb[base + j * 64]; }
  ql[idx] = (bf16)(sq * (1.f / 16.f));
  kl[idx] = (bf16)(sk * (1.f / 16.f));
}

// ---------------- generic batched NT GEMM (256-class), optional epilogues ----------
__global__ void k_gemm_nt(const bf16* __restrict__ A, int lda, long aBS,
                          const bf16* __restrict__ Bm, int ldb, long bBS,
                          int ksteps, float scale, int Nn,
                          float* outF, long fBS, bf16* outB, long oBS,
                          bf16* outT, int ldT, long tBS) {
  int bh = blockIdx.z;
  int row0 = blockIdx.y * 16, col0 = blockIdx.x * 64;
  const bf16* Ab = A + (long)bh * aBS;
  const bf16* Bb = Bm + (long)bh * bBS;
  v8f acc[4] = {};
  for (int ks = 0; ks < ksteps; ++ks) {
    int k0 = ks * 32;
    v16bf af = ld_frag(Ab, lda, row0, k0);
#pragma unroll
    for (int t = 0; t < 4; ++t)
      acc[t] = wmma_bf(af, ld_frag(Bb, ldb, col0 + t * 16, k0), acc[t]);
  }
  int lane = threadIdx.x & 31, c = lane & 15, hf = lane >> 4;
#pragma unroll
  for (int t = 0; t < 4; ++t) {
    int col = col0 + t * 16 + c;
#pragma unroll
    for (int v = 0; v < 8; ++v) {
      int row = row0 + hf * 8 + v;
      float val = acc[t][v] * scale;
      if (outF) outF[(long)bh * fBS + (long)row * Nn + col] = val;
      if (outB) outB[(long)bh * oBS + (long)row * Nn + col] = (bf16)val;
      if (outT) outT[(long)bh * tBS + (long)col * ldT + row] = (bf16)val;
    }
  }
}

// ---------------- row softmax over 256 cols -> bf16 ----------------
__global__ void k_softmax256(const float* __restrict__ in, bf16* __restrict__ outb) {
  __shared__ float red[256];
  long row = blockIdx.x;
  int  t   = threadIdx.x;
  float v = in[row * 256 + t];
  red[t] = v; __syncthreads();
  for (int off = 128; off > 0; off >>= 1) {
    if (t < off) red[t] = fmaxf(red[t], red[t + off]);
    __syncthreads();
  }
  float m = red[0]; __syncthreads();
  float e = __expf(v - m);
  red[t] = e; __syncthreads();
  for (int off = 128; off > 0; off >>= 1) {
    if (t < off) red[t] += red[t + off];
    __syncthreads();
  }
  outb[row * 256 + t] = (bf16)(e / red[0]);
}

// ---------------- pinv init helpers ----------------
__global__ void k_colsum(const bf16* __restrict__ mat, float* __restrict__ cs) {
  int idx = blockIdx.x * blockDim.x + threadIdx.x;
  if (idx >= BH_ * M_) return;
  int bh = idx >> 8, j = idx & 255;
  const bf16* b = mat + ((long)bh << 16) + j;
  float s = 0.f;
  for (int i = 0; i < 256; ++i) s += (float)b[i * 256];
  cs[idx] = s;
}

__global__ void k_scale(const float* __restrict__ cs, float* __restrict__ sc) {
  __shared__ float red[256];
  int t = threadIdx.x;
  float m = -1e30f;
  for (int i = t; i < BH_ * M_; i += 256) m = fmaxf(m, cs[i]);
  red[t] = m; __syncthreads();
  for (int off = 128; off > 0; off >>= 1) {
    if (t < off) red[t] = fmaxf(red[t], red[t + off]);
    __syncthreads();
  }
  if (t == 0) sc[0] = 1.0f / red[0];   // row-sum max of a softmax == 1
}

// z0 = mat^T * scale; write z and z^T
__global__ void k_zinit(const bf16* __restrict__ mat, const float* __restrict__ sc,
                        bf16* __restrict__ z, bf16* __restrict__ zT) {
  long idx = (long)blockIdx.x * blockDim.x + threadIdx.x;
  if (idx >= (long)BH_ << 16) return;
  int j = idx & 255;
  long r = idx >> 8;
  int i = (int)(r & 255), bh = (int)(r >> 8);
  float val = (float)mat[idx] * sc[0];            // = z[j][i]
  zT[idx] = (bf16)val;                            // zT[i][j]
  z[((long)bh << 16) + (long)j * 256 + i] = (bf16)val;
}

// uT[j][k] = alpha*(j==k) - in[k][j]   (transposed store, bf16)
__global__ void k_ueps(const float* __restrict__ in, bf16* __restrict__ uT, float alpha) {
  long idx = (long)blockIdx.x * blockDim.x + threadIdx.x;
  if (idx >= (long)BH_ << 16) return;
  int k = idx & 255;
  long r = idx >> 8;
  int j = (int)(r & 255), bh = (int)(r >> 8);
  float v = -in[((long)bh << 16) + (long)k * 256 + j];
  if (j == k) v += alpha;
  uT[idx] = (bf16)v;
}

// ---------------- online-softmax helper (per-wave) ----------------
__device__ __forceinline__ void online_update(v8f s0, v8f s1, v8f* o,
                                              float* rmax, float* rsum, bf16* plds) {
  int lane = threadIdx.x & 31, cIn = lane & 15, hf = lane >> 4;
#pragma unroll
  for (int v = 0; v < 8; ++v) {
    float tm = fmaxf(s0[v], s1[v]);
    tm = fmaxf(tm, __shfl_xor(tm, 1, 16));
    tm = fmaxf(tm, __shfl_xor(tm, 2, 16));
    tm = fmaxf(tm, __shfl_xor(tm, 4, 16));
    tm = fmaxf(tm, __shfl_xor(tm, 8, 16));
    float nm = fmaxf(rmax[v], tm);
    float al = __expf(rmax[v] - nm);
    float p0 = __expf(s0[v] - nm), p1 = __expf(s1[v] - nm);
    float ps = p0 + p1;
    ps += __shfl_xor(ps, 1, 16);
    ps += __shfl_xor(ps, 2, 16);
    ps += __shfl_xor(ps, 4, 16);
    ps += __shfl_xor(ps, 8, 16);
    rsum[v] = rsum[v] * al + ps;
    rmax[v] = nm;
    int r = hf * 8 + v;
    plds[r * 32 + cIn]      = (bf16)p0;
    plds[r * 32 + 16 + cIn] = (bf16)p1;
#pragma unroll
    for (int t = 0; t < 4; ++t) o[t][v] *= al;
  }
  __syncthreads();
}

// ---------------- fused: w1 = softmax(q_l k^T) @ v  (flash over n=4096) ------------
// writes w1 transposed: w1T[bh][dh][m]
__global__ void k_attn3v(const bf16* __restrict__ ql, const bf16* __restrict__ kb,
                         const bf16* __restrict__ vT, bf16* __restrict__ w1T) {
  int bh = blockIdx.y, m0 = blockIdx.x * 16;
  const bf16* qlb = ql + (long)bh * M_ * DH_;
  const bf16* kbb = kb + ((long)bh << 12) * 64;
  const bf16* vTb = vT + (long)bh * 64 * 4096;
  v16bf aq0 = ld_frag(qlb, 64, m0, 0);
  v16bf aq1 = ld_frag(qlb, 64, m0, 32);
  v8f o[4] = {};
  float rmax[8], rsum[8];
#pragma unroll
  for (int v = 0; v < 8; ++v) { rmax[v] = -1e30f; rsum[v] = 0.f; }
  __shared__ __align__(16) bf16 plds[16 * 32];
  int lane = threadIdx.x & 31, cIn = lane & 15, hf = lane >> 4;
  for (int nb = 0; nb < 4096; nb += 32) {
    v8f s0 = {}, s1 = {};
    s0 = wmma_bf(aq0, ld_frag(kbb, 64, nb, 0), s0);
    s0 = wmma_bf(aq1, ld_frag(kbb, 64, nb, 32), s0);
    s1 = wmma_bf(aq0, ld_frag(kbb, 64, nb + 16, 0), s1);
    s1 = wmma_bf(aq1, ld_frag(kbb, 64, nb + 16, 32), s1);
    online_update(s0, s1, o, rmax, rsum, plds);
    v16bf ap = ld_frag(plds, 32, 0, 0);
#pragma unroll
    for (int t = 0; t < 4; ++t)
      o[t] = wmma_bf(ap, ld_frag(vTb, 4096, t * 16, nb), o[t]);
    __syncthreads();
  }
#pragma unroll
  for (int t = 0; t < 4; ++t) {
    int dh = t * 16 + cIn;
#pragma unroll
    for (int v = 0; v < 8; ++v) {
      int row = m0 + hf * 8 + v;
      w1T[((long)bh * 64 + dh) * 256 + row] = (bf16)(o[t][v] / rsum[v]);
    }
  }
}

// ---------------- depthwise conv residual over sequence ----------------
__global__ void k_conv(const bf16* __restrict__ vb, const float* __restrict__ ker,
                       float* __restrict__ res) {
  long idx = (long)blockIdx.x * blockDim.x + threadIdx.x;
  if (idx >= (long)BH_ * N_ * DH_) return;
  int dh = idx & 63;
  long r = idx >> 6;
  int tok = (int)(r & 4095), bh = (int)(r >> 12);
  int hh = bh & 7;
  float s = 0.f;
#pragma unroll
  for (int j = 0; j < KS_; ++j) {
    int tt = tok + j - (KS_ / 2);
    if (tt >= 0 && tt < N_)
      s += (float)vb[((long)bh << 18) + ((long)tt << 6) + dh] * ker[hh * KS_ + j];
  }
  res[idx] = s;
}

// ---------------- fused: out_h = softmax(q kl^T) @ t + res, merge heads ------------
__global__ void k_attn1_out(const bf16* __restrict__ qb, const bf16* __restrict__ kl,
                            const bf16* __restrict__ tT, const float* __restrict__ res,
                            bf16* __restrict__ merged) {
  int bh = blockIdx.y, n0 = blockIdx.x * 16;
  const bf16* qbb = qb + ((long)bh << 12) * 64;
  const bf16* klb = kl + (long)bh * M_ * DH_;
  const bf16* tTb = tT + (long)bh * 64 * 256;
  v16bf aq0 = ld_frag(qbb, 64, n0, 0);
  v16bf aq1 = ld_frag(qbb, 64, n0, 32);
  v8f o[4] = {};
  float rmax[8], rsum[8];
#pragma unroll
  for (int v = 0; v < 8; ++v) { rmax[v] = -1e30f; rsum[v] = 0.f; }
  __shared__ __align__(16) bf16 plds[16 * 32];
  int lane = threadIdx.x & 31, cIn = lane & 15, hf = lane >> 4;
  for (int mb = 0; mb < 256; mb += 32) {
    v8f s0 = {}, s1 = {};
    s0 = wmma_bf(aq0, ld_frag(klb, 64, mb, 0), s0);
    s0 = wmma_bf(aq1, ld_frag(klb, 64, mb, 32), s0);
    s1 = wmma_bf(aq0, ld_frag(klb, 64, mb + 16, 0), s1);
    s1 = wmma_bf(aq1, ld_frag(klb, 64, mb + 16, 32), s1);
    online_update(s0, s1, o, rmax, rsum, plds);
    v16bf ap = ld_frag(plds, 32, 0, 0);
#pragma unroll
    for (int t = 0; t < 4; ++t)
      o[t] = wmma_bf(ap, ld_frag(tTb, 256, t * 16, mb), o[t]);
    __syncthreads();
  }
  int b = bh >> 3, hh = bh & 7;
#pragma unroll
  for (int t = 0; t < 4; ++t) {
    int dh = t * 16 + cIn;
#pragma unroll
    for (int v = 0; v < 8; ++v) {
      int tok = n0 + hf * 8 + v;
      float val = o[t][v] / rsum[v] + res[(((long)bh << 12) + tok) * 64 + dh];
      merged[((long)(b * N_ + tok)) * D_ + hh * 64 + dh] = (bf16)val;
    }
  }
}

// ---------------- out projection + bias + outer residual ----------------
__global__ void k_outproj(const bf16* __restrict__ mg, const bf16* __restrict__ wT,
                          const float* __restrict__ x, const float* __restrict__ bo,
                          float* __restrict__ out) {
  int row0 = blockIdx.y * 16, col0 = blockIdx.x * 64;
  v8f acc[4] = {};
  for (int ks = 0; ks < 16; ++ks) {
    int k0 = ks * 32;
    v16bf af = ld_frag(mg, D_, row0, k0);
#pragma unroll
    for (int t = 0; t < 4; ++t)
      acc[t] = wmma_bf(af, ld_frag(wT, D_, col0 + t * 16, k0), acc[t]);
  }
  int lane = threadIdx.x & 31, c = lane & 15, hf = lane >> 4;
#pragma unroll
  for (int t = 0; t < 4; ++t) {
    int col = col0 + t * 16 + c;
#pragma unroll
    for (int v = 0; v < 8; ++v) {
      long row = row0 + hf * 8 + v;
      out[row * D_ + col] = x[row * D_ + col] + bo[col] + acc[t][v];
    }
  }
}

extern "C" void kernel_launch(void* const* d_in, const int* in_sizes, int n_in,
                              void* d_out, int out_size, void* d_ws, size_t ws_size,
                              hipStream_t stream) {
  (void)in_sizes; (void)n_in; (void)out_size; (void)ws_size;
  const float* x     = (const float*)d_in[0];
  const float* wqkv  = (const float*)d_in[1];
  const float* wout  = (const float*)d_in[2];
  const float* bo    = (const float*)d_in[3];
  const float* rker  = (const float*)d_in[4];
  const float* gamma = (const float*)d_in[5];
  const float* beta  = (const float*)d_in[6];
  float* out = (float*)d_out;

  char* p = (char*)d_ws;
  auto carve = [&](size_t bytes) -> char* {
    char* r = p;
    p += (bytes + 255) & ~(size_t)255;
    return r;
  };
  bf16*  xn     = (bf16*)carve((size_t)NT_ * D_ * 2);
  bf16*  wqkvT  = (bf16*)carve((size_t)1536 * D_ * 2);
  bf16*  woutT  = (bf16*)carve((size_t)D_ * D_ * 2);
  bf16*  qb     = (bf16*)carve((size_t)BH_ * N_ * DH_ * 2);
  bf16*  kb     = (bf16*)carve((size_t)BH_ * N_ * DH_ * 2);
  bf16*  vb     = (bf16*)carve((size_t)BH_ * N_ * DH_ * 2);
  bf16*  vT     = (bf16*)carve((size_t)BH_ * N_ * DH_ * 2);
  bf16*  ql     = (bf16*)carve((size_t)BH_ * M_ * DH_ * 2);
  bf16*  kl     = (bf16*)carve((size_t)BH_ * M_ * DH_ * 2);
  float* scratF = (float*)carve((size_t)BH_ * M_ * M_ * 4);   // logits2 / mz_f32
  bf16*  matb   = (bf16*)carve((size_t)BH_ * M_ * M_ * 2);
  bf16*  zA     = (bf16*)carve((size_t)BH_ * M_ * M_ * 2);
  bf16*  zAT    = (bf16*)carve((size_t)BH_ * M_ * M_ * 2);
  bf16*  zB     = (bf16*)carve((size_t)BH_ * M_ * M_ * 2);
  bf16*  zBT    = (bf16*)carve((size_t)BH_ * M_ * M_ * 2);
  bf16*  mzb    = (bf16*)carve((size_t)BH_ * M_ * M_ * 2);
  float* uf     = (float*)carve((size_t)BH_ * M_ * M_ * 4);
  bf16*  uT     = (bf16*)carve((size_t)BH_ * M_ * M_ * 2);
  bf16*  w1T    = (bf16*)carve((size_t)BH_ * DH_ * M_ * 2);
  bf16*  tT     = (bf16*)carve((size_t)BH_ * DH_ * M_ * 2);
  float* res    = (float*)carve((size_t)BH_ * N_ * DH_ * 4);
  bf16*  merged = (bf16*)carve((size_t)NT_ * D_ * 2);
  float* colsum = (float*)carve((size_t)BH_ * M_ * 4);
  float* sc     = (float*)carve(256);

  const long MM = (long)M_ * M_;     // 65536
  const long MD = (long)M_ * DH_;    // 16384

  // 1. layernorm + cast
  k_ln<<<NT_, 256, 0, stream>>>(x, gamma, beta, xn);
  // 2. weight transpose + cast
  k_prepw<<<(512 * 1536 + 512 * 512 + 255) / 256, 256, 0, stream>>>(wqkv, wout, wqkvT, woutT);
  // 3. qkv GEMM
  k_qkv<<<dim3(1536 / 64, NT_ / 16), 32, 0, stream>>>(xn, wqkvT, qb, kb, vb, vT);
  // 4. landmarks
  k_landmark<<<(BH_ * M_ * DH_ + 255) / 256, 256, 0, stream>>>(qb, kb, ql, kl);
  // 5. attn2 logits (q_l @ k_l^T), K=64
  k_gemm_nt<<<dim3(M_ / 64, M_ / 16, BH_), 32, 0, stream>>>(
      ql, 64, MD, kl, 64, MD, 2, 1.0f, M_, scratF, MM, nullptr, 0, nullptr, 0, 0);
  // 6. softmax -> mat (bf16)
  k_softmax256<<<BH_ * M_, 256, 0, stream>>>(scratF, matb);
  // 7-8. global scale for pinv init
  k_colsum<<<(BH_ * M_ + 255) / 256, 256, 0, stream>>>(matb, colsum);
  k_scale<<<1, 256, 0, stream>>>(colsum, sc);
  // 9. z0 = mat^T * scale
  k_zinit<<<(int)(((long)BH_ << 16) / 256), 256, 0, stream>>>(matb, sc, zA, zAT);

  // 10. Newton-Schulz pinv iterations
  bf16 *zin = zA, *zinT = zAT, *zout = zB, *zoutT = zBT;
  for (int it = 0; it < 6; ++it) {
    // mz = mat @ z
    k_gemm_nt<<<dim3(M_ / 64, M_ / 16, BH_), 32, 0, stream>>>(
        matb, M_, MM, zinT, M_, MM, 8, 1.0f, M_, scratF, MM, mzb, MM, nullptr, 0, 0);
    k_ueps<<<(int)(((long)BH_ << 16) / 256), 256, 0, stream>>>(scratF, uT, 7.0f);
    // u2 = mz @ (7I - mz)
    k_gemm_nt<<<dim3(M_ / 64, M_ / 16, BH_), 32, 0, stream>>>(
        mzb, M_, MM, uT, M_, MM, 8, 1.0f, M_, uf, MM, nullptr, 0, nullptr, 0, 0);
    k_ueps<<<(int)(((long)BH_ << 16) / 256), 256, 0, stream>>>(uf, uT, 15.0f);
    // u4 = mz @ (15I - u2)
    k_gemm_nt<<<dim3(M_ / 64, M_ / 16, BH_), 32, 0, stream>>>(
        mzb, M_, MM, uT, M_, MM, 8, 1.0f, M_, uf, MM, nullptr, 0, nullptr, 0, 0);
    k_ueps<<<(int)(((long)BH_ << 16) / 256), 256, 0, stream>>>(uf, uT, 13.0f);
    // z = 0.25 * z @ (13I - u4)   (write z and z^T, ping-pong)
    k_gemm_nt<<<dim3(M_ / 64, M_ / 16, BH_), 32, 0, stream>>>(
        zin, M_, MM, uT, M_, MM, 8, 0.25f, M_, nullptr, 0, zout, MM, zoutT, M_, MM);
    bf16* s1 = zin; zin = zout; zout = s1;
    bf16* s2 = zinT; zinT = zoutT; zoutT = s2;
  }

  // 11. w1 = softmax(q_l k^T) @ v   (flash over n)
  k_attn3v<<<dim3(M_ / 16, BH_), 32, 0, stream>>>(ql, kb, vT, w1T);
  // 12. t = z @ w1  (write t^T)
  k_gemm_nt<<<dim3(1, M_ / 16, BH_), 32, 0, stream>>>(
      zin, M_, MM, w1T, M_, MD, 8, 1.0f, DH_, nullptr, 0, nullptr, 0, tT, M_, MD);
  // 13. depthwise conv residual
  k_conv<<<(int)(((long)BH_ * N_ * DH_) / 256), 256, 0, stream>>>(vb, rker, res);
  // 14. out_h = softmax(q kl^T) @ t + res, merge heads
  k_attn1_out<<<dim3(N_ / 16, BH_), 32, 0, stream>>>(qb, kl, tT, res, merged);
  // 15. final projection + bias + outer residual
  k_outproj<<<dim3(D_ / 64, NT_ / 16), 32, 0, stream>>>(merged, woutT, x, bo, out);
}
